// Actor_Net_88321707475763
// MI455X (gfx1250) — matrix operand order repaired and checked by reading, not compiled, not run on hardware
//
#include <hip/hip_runtime.h>
#include <hip/hip_fp16.h>

typedef __attribute__((ext_vector_type(16))) _Float16 v16h;
typedef __attribute__((ext_vector_type(8)))  _Float16 v8h;
typedef __attribute__((ext_vector_type(8)))  float    v8f;

#define M_TOTAL    236672      // 32 * 86 * 86
#define OUT_HALF   946688      // M_TOTAL * 4
#define NBLOCKS    1849        // M_TOTAL / 128
#define ROWS_WG    128
#define SA         168         // feat row stride in f16 (K padded 144->160, +8 pad)
#define SH         264         // hidden row stride in f16 (256 + 8 pad)

// LDS byte offsets (dynamic shared memory)
#define OFF_FEAT   0
#define OFF_HA     43008       // 128*168*2
#define OFF_HB     110592      // + 128*264*2
#define OFF_W      178176      // + 128*264*2
#define SMEM_BYTES 309248      // + 131072 (<= 320KB WGP LDS)

// workspace (dword offsets) for swizzled f16 weights
#define WS_W1 0                // Kp=160, N=256 -> 20480 dwords
#define WS_W2 20480            // K=256,  N=256 -> 32768 dwords
#define WS_W3 53248            // K=256,  N=128 -> 16384 dwords
#define WS_WH 69632            // K=128,  N=16  -> 1024 dwords
#define WS_TOTAL 70656

// ---------------------------------------------------------------------------
// Weight prep: f32 -> f16, swizzled so a B fragment (8 VGPRs/lane) is one
// contiguous 32B read:  dword idx = ((k>>4)*N + n)*8 + ((k>>1)&7)
// ---------------------------------------------------------------------------
__global__ void prep_weights_kernel(const float* __restrict__ W1,
                                    const float* __restrict__ W2,
                                    const float* __restrict__ W3,
                                    const float* __restrict__ Wm,
                                    const float* __restrict__ Wl,
                                    unsigned* __restrict__ ws) {
  unsigned d = blockIdx.x * blockDim.x + threadIdx.x;
  if (d >= (unsigned)WS_TOTAL) return;
  const float* src = nullptr;
  unsigned loc, N, Ksrc; int mode = 0;
  if (d < (unsigned)WS_W2)      { loc = d;         src = W1; N = 256; Ksrc = 144; }
  else if (d < (unsigned)WS_W3) { loc = d - WS_W2; src = W2; N = 256; Ksrc = 256; }
  else if (d < (unsigned)WS_WH) { loc = d - WS_W3; src = W3; N = 128; Ksrc = 256; }
  else                          { loc = d - WS_WH; N = 16; Ksrc = 128; mode = 1; }
  unsigned v = loc & 7u, t = loc >> 3;
  unsigned n = t % N, g = t / N;
  unsigned k0 = g * 16u + v * 2u;
  float a0, a1;
  if (!mode) {
    a0 = (k0     < Ksrc) ? src[k0 * N + n]       : 0.f;
    a1 = (k0 + 1 < Ksrc) ? src[(k0 + 1) * N + n] : 0.f;
  } else {  // heads: cols 0-3 = Wm, 4-7 = Wl, 8-15 = 0
    a0 = (n < 4u) ? Wm[k0 * 4 + n] : ((n < 8u) ? Wl[k0 * 4 + (n - 4u)] : 0.f);
    a1 = (n < 4u) ? Wm[(k0 + 1) * 4 + n] : ((n < 8u) ? Wl[(k0 + 1) * 4 + (n - 4u)] : 0.f);
  }
  union { _Float16 h[2]; unsigned u; } pk;
  pk.h[0] = (_Float16)a0; pk.h[1] = (_Float16)a1;
  ws[d] = pk.u;
}

// ---------------------------------------------------------------------------
// Async weight staging into LDS (CDNA5 GLOBAL_LOAD_ASYNC_TO_LDS, ASYNCcnt)
// ---------------------------------------------------------------------------
__device__ inline void stage_issue(unsigned lds_off, const unsigned* gsrc,
                                   int ndwords, int tid) {
  for (int i = tid * 4; i < ndwords; i += 1024) {
    unsigned lo = lds_off + (unsigned)i * 4u;
    unsigned long long ga = (unsigned long long)(gsrc + i);
    asm volatile("global_load_async_to_lds_b128 %0, %1, off"
                 :: "v"(lo), "v"(ga) : "memory");
  }
}
__device__ inline void async_wait() {
  asm volatile("s_wait_asynccnt 0" ::: "memory");
}

// ---------------------------------------------------------------------------
// Fragment loaders (wave32 WMMA layouts, CDNA5 ISA §7.12.2)
// ---------------------------------------------------------------------------
__device__ inline v16h ld2(const _Float16* p0, const _Float16* p1) {
  union { v16h v; v8h h[2]; } u;
  u.h[0] = *(const v8h*)p0;
  u.h[1] = *(const v8h*)p1;
  return u.v;
}

// A 16x32: lanes 0-15 -> M=0..15 kb=0 ; lanes 16-31 -> kb=16; V0-7=K[kb..kb+15]
__device__ inline v16h load_A(const _Float16* As, int sa, int rowBase, int lane, int kstep) {
  int row = rowBase + (lane & 15);
  int kb  = kstep * 32 + ((lane >> 4) << 4);
  const _Float16* p = As + row * sa + kb;
  return ld2(p, p + 8);
}

// B 32x16 from swizzled dwords: lane n=l&15, k16=l>>4, 8 contiguous dwords
__device__ inline v16h load_B(const unsigned* Bw, int N, int lane, int kstep, int ntile) {
  int n = (ntile << 4) + (lane & 15);
  int g = kstep * 2 + (lane >> 4);
  const unsigned* p = Bw + (((unsigned)(g * N + n)) << 3);
  return ld2((const _Float16*)p, (const _Float16*)(p + 4));
}

// ---------------------------------------------------------------------------
// One wave computes a 16 x (NT*16) GEMM slab, K = KS*32, epilogue -> LDS f16
// ---------------------------------------------------------------------------
template <int KS, int NT, bool RELU>
__device__ inline void wave_gemm(const _Float16* As, int sa, int rowBase,
                                 const unsigned* Bw, int N,
                                 const float* __restrict__ bias,
                                 _Float16* Dst, int sh, int lane) {
  v16h afr[KS];
#pragma unroll
  for (int k = 0; k < KS; ++k) afr[k] = load_A(As, sa, rowBase, lane, k);

  const int ncol = lane & 15;
  const int drow = rowBase + ((lane >> 4) << 3);
#pragma unroll
  for (int nt = 0; nt < NT; ++nt) {
    v8f acc = {};
#pragma unroll
    for (int k = 0; k < KS; ++k) {
      v16h bf = load_B(Bw, N, lane, k, nt);
      acc = __builtin_amdgcn_wmma_f32_16x16x32_f16(false, afr[k], false, bf,
                                                   (short)0, acc, false, false);
    }
    float bv = bias[nt * 16 + ncol];
#pragma unroll
    for (int d = 0; d < 8; ++d) {
      float v = acc[d] + bv;
      if (RELU) v = fmaxf(v, 0.f);
      Dst[(drow + d) * sh + (nt * 16 + ncol)] = (_Float16)v;
    }
  }
}

// heads: 16x128 @ 128x16 (cols 0-3 mean, 4-7 log_std), writes global out
__device__ inline void wave_heads(const _Float16* As, int sa, int rowBase,
                                  const unsigned* Bw,
                                  const float* __restrict__ bm,
                                  const float* __restrict__ bl,
                                  float* __restrict__ out, long mbase, int lane) {
  v8f acc = {};
#pragma unroll
  for (int k = 0; k < 4; ++k) {
    v16h a = load_A(As, sa, rowBase, lane, k);
    v16h b = load_B(Bw, 16, lane, k, 0);
    acc = __builtin_amdgcn_wmma_f32_16x16x32_f16(false, a, false, b,
                                                 (short)0, acc, false, false);
  }
  const int col = lane & 15;
  const int drow = rowBase + ((lane >> 4) << 3);
#pragma unroll
  for (int d = 0; d < 8; ++d) {
    long m = mbase + drow + d;
    float v = acc[d];
    if (col < 4)      out[m * 4 + col] = v + bm[col];
    else if (col < 8) out[(long)OUT_HALF + m * 4 + (col - 4)] = v + bl[col - 4];
  }
}

// ---------------------------------------------------------------------------
// Fused kernel: patches + rank transform + 4 GEMM layers, all in LDS
// ---------------------------------------------------------------------------
__global__ __launch_bounds__(256)
void actor_fused_kernel(const float* __restrict__ x,
                        const float* __restrict__ b1, const float* __restrict__ b2,
                        const float* __restrict__ b3, const float* __restrict__ bm,
                        const float* __restrict__ bl,
                        const unsigned* __restrict__ ws, float* __restrict__ out) {
  extern __shared__ char smem[];
  _Float16* feat = (_Float16*)(smem + OFF_FEAT);
  _Float16* hA   = (_Float16*)(smem + OFF_HA);
  _Float16* hB   = (_Float16*)(smem + OFF_HB);
  unsigned* wlds = (unsigned*)(smem + OFF_W);
  const unsigned wloff = (unsigned)(unsigned long long)(void*)wlds;  // LDS byte addr

  const int tid = threadIdx.x;
  const int lane = tid & 31, wave = tid >> 5, rowBase = wave * 16;

  // kick off W1 staging while we build the feature tile
  stage_issue(wloff, ws + WS_W1, 20480, tid);
  __builtin_prefetch(ws + WS_W2 + ((tid & 255) << 7), 0, 1);  // warm W2 in L2

  // ---- feature tile: 128 rows x 144 (padded 160) f16 ----
  {
    float code = x[0];
    float it = fmodf(code, 100.0f);
    int iti = (int)it;
    int offh = iti % 3, offw = (iti / 3) % 3;
    float xfix = (code - it) * 0.01f;

    int r = tid & 127, half = tid >> 7;                 // 2 threads / row
    long m = (long)blockIdx.x * ROWS_WG + r;
    int b = (int)(m / 7396);
    int rem = (int)(m - (long)b * 7396);
    int ph = rem / 86, pw = rem - ph * 86;

    float vals[9][8];
    for (int p = 0; p < 9; ++p) {
      int i = p / 3, j = p - 3 * i;
      int hh = ph * 3 + i - offh;
      int ww = pw * 3 + j - offw;
      float4 a = make_float4(0.f, 0.f, 0.f, 0.f);
      float4 c = make_float4(0.f, 0.f, 0.f, 0.f);
      if ((unsigned)hh < 256u && (unsigned)ww < 256u) {
        const float* sp = x + ((((long)b * 256 + hh) * 256 + ww) * 16 + half * 8);
        a = *(const float4*)sp;
        c = *(const float4*)(sp + 4);
        if (half == 0 && b == 0 && hh == 0 && ww == 0) a.x = xfix;  // decode slot
      }
      vals[p][0] = a.x; vals[p][1] = a.y; vals[p][2] = a.z; vals[p][3] = a.w;
      vals[p][4] = c.x; vals[p][5] = c.y; vals[p][6] = c.z; vals[p][7] = c.w;
    }
    if (half == 0) {  // rank-by-first-appearance on channel 0 (9 values)
      float t0[9];
      for (int p = 0; p < 9; ++p) {
        int fo = p;
        for (int q = 0; q < p; ++q)
          if (vals[q][0] == vals[p][0]) { fo = q; break; }
        int rk = 0;
        for (int q = 0; q < fo; ++q) {
          int fq = q;
          for (int s = 0; s < q; ++s)
            if (vals[s][0] == vals[q][0]) { fq = s; break; }
          rk += (fq == q);
        }
        t0[p] = (vals[p][0] == 0.f) ? 0.f : (float)(rk + 1);
      }
      for (int p = 0; p < 9; ++p) vals[p][0] = t0[p];
    }
    _Float16* frow = feat + r * SA + half * 8;
    for (int p = 0; p < 9; ++p)
      for (int c = 0; c < 8; ++c)
        frow[p * 16 + c] = (_Float16)vals[p][c];
    for (int c = 0; c < 8; ++c)                          // zero K-pad 144..159
      feat[r * SA + 144 + half * 8 + c] = (_Float16)0.f;
  }
  async_wait();
  __syncthreads();

  // L1: feat(128x160) @ W1(160x256) -> hA, relu
  wave_gemm<5, 16, true>(feat, SA, rowBase, wlds, 256, b1, hA, SH, lane);
  __syncthreads();
  stage_issue(wloff, ws + WS_W2, 32768, tid);
  __builtin_prefetch(ws + WS_W3 + ((tid & 255) << 6), 0, 1);
  async_wait();
  __syncthreads();

  // L2: hA(128x256) @ W2(256x256) -> hB, relu
  wave_gemm<8, 16, true>(hA, SH, rowBase, wlds, 256, b2, hB, SH, lane);
  __syncthreads();
  stage_issue(wloff, ws + WS_W3, 16384, tid);
  async_wait();
  __syncthreads();

  // L3: hB(128x256) @ W3(256x128) -> hA (no relu)
  wave_gemm<8, 8, false>(hB, SH, rowBase, wlds, 128, b3, hA, SH, lane);
  __syncthreads();
  stage_issue(wloff, ws + WS_WH, 1024, tid);
  async_wait();
  __syncthreads();

  // heads: hA(128x128) @ [Wm|Wl](128x16) -> global mean/log_std
  wave_heads(hA, SH, rowBase, wlds, bm, bl, out, (long)blockIdx.x * ROWS_WG, lane);
}

// ---------------------------------------------------------------------------
extern "C" void kernel_launch(void* const* d_in, const int* in_sizes, int n_in,
                              void* d_out, int out_size, void* d_ws, size_t ws_size,
                              hipStream_t stream) {
  const float* x  = (const float*)d_in[0];
  const float* W1 = (const float*)d_in[1];
  const float* b1 = (const float*)d_in[2];
  const float* W2 = (const float*)d_in[3];
  const float* b2 = (const float*)d_in[4];
  const float* W3 = (const float*)d_in[5];
  const float* b3 = (const float*)d_in[6];
  const float* Wm = (const float*)d_in[7];
  const float* bm = (const float*)d_in[8];
  const float* Wl = (const float*)d_in[9];
  const float* bl = (const float*)d_in[10];
  unsigned* ws = (unsigned*)d_ws;
  float* out = (float*)d_out;

  prep_weights_kernel<<<(WS_TOTAL + 255) / 256, 256, 0, stream>>>(W1, W2, W3, Wm, Wl, ws);
  actor_fused_kernel<<<NBLOCKS, 256, SMEM_BYTES, stream>>>(x, b1, b2, b3, bm, bl, ws, out);
}
